// VQ_Projector_46823733461245
// MI455X (gfx1250) — matrix operand order repaired and checked by reading, not compiled, not run on hardware
//
#include <hip/hip_runtime.h>

// ---------------------------------------------------------------------------
// VQ_Projector for MI455X (gfx1250).
//   - encoder GEMMs + VQ scoring via v_wmma_f32_16x16x32_bf16 (f32 accumulate)
//   - global->LDS staging via GLOBAL_LOAD_ASYNC_TO_LDS_B128 (ASYNCcnt) with
//     ping-pong double buffering; falls back to manual copies if the builtin
//     is unavailable.
//   - bf16 operand conversion done once in bandwidth-light prep kernels.
//   - decoder in the reference is dead code (result discarded) -> skipped.
// ---------------------------------------------------------------------------

typedef __attribute__((ext_vector_type(16))) __bf16 v16bf;
typedef __attribute__((ext_vector_type(8)))  float  v8f;
typedef __attribute__((ext_vector_type(4)))  int    vi4;

union AFragU { v16bf v; unsigned u[8]; };

#if defined(__HIP_DEVICE_COMPILE__) && __has_builtin(__builtin_amdgcn_global_load_async_to_lds_b128)
#define HAVE_ASYNC 1
#else
#define HAVE_ASYNC 0
#endif

__device__ __forceinline__ void async_cp16(const void* g, void* l) {
#if HAVE_ASYNC
  __builtin_amdgcn_global_load_async_to_lds_b128((vi4*)g, (vi4*)l, 0, 0);
#else
  *(uint4*)l = *(const uint4*)g;
#endif
}

__device__ __forceinline__ void wait_async0() {
#if HAVE_ASYNC
#if __has_builtin(__builtin_amdgcn_s_wait_asynccnt)
  __builtin_amdgcn_s_wait_asynccnt(0);
#else
  asm volatile("s_wait_asynccnt 0x0" ::: "memory");
#endif
#endif
}

// 16-bit A-matrix 16x32 layout: lanes 0-15 hold M=lane, K in {0..7,16..23};
// lanes 16-31 hold K in {8..15,24..31}; VGPR e holds the K pair below.
__device__ __forceinline__ int amapA(int e, int hi) {
  return 2 * e + ((e >= 4) ? 8 : 0) + hi * 8;
}
// 16-bit B-matrix 32x16 layout: lane = N, lanes 0-15 K in {0..7,16..23}, lanes 16-31 K in {8..15,24..31}.
__device__ __forceinline__ int amapB(int e, int hi) {
  return 2 * e + ((e >= 4) ? 8 : 0) + hi * 8;
}

// ---------------------------------------------------------------------------
// bf16 WMMA GEMM, 128x128 block tile, async double-buffered LDS staging.
//   A : bf16 row-major [M,K] (K multiple of 32, rows 16B aligned)
//   Bt: bf16 [N][GSB] = B^T with padded row stride GSB (16B-aligned rows)
//   C : bf16 (GEMM1) or f32 (GEMM2), = A@B + bias
// 8 waves: wave tile 64(M) x 32(N) = 4x2 accumulators of 16x16.
// ---------------------------------------------------------------------------
template <bool C_BF16>
__global__ __launch_bounds__(256) void gemm_wmma_async(
    const __bf16* __restrict__ A, const __bf16* __restrict__ Bt,
    const float* __restrict__ bias, void* __restrict__ Craw,
    int M, int K, int N, int GSB)
{
  constexpr int LS = 40;                          // LDS K-stride: 80 B rows
  __shared__ __align__(16) __bf16 sA[2][128 * LS];
  __shared__ __align__(16) __bf16 sB[2][128 * LS];

  const int tid  = threadIdx.x;
  const int lane = tid & 31;
  const int wv   = tid >> 5;
  const int l16  = lane & 15;
  const int hi   = lane >> 4;
  const int wm   = (wv >> 2) * 64;
  const int wn   = (wv & 3) * 32;
  const int bm   = blockIdx.x * 128;
  const int bn   = blockIdx.y * 128;

  // staging: thread t copies two 16B chunks of row (t>>1), byte offset (t&1)*32
  const int smr = tid >> 1;
  const int sco = (tid & 1) * 32;

  auto stage = [&](int buf, int k0) {
    {
      const char* g = (const char*)(A + (size_t)(bm + smr) * K + k0) + sco;
      char* l = (char*)&sA[buf][smr * LS] + sco;
      async_cp16(g, l);
      async_cp16(g + 16, l + 16);
    }
    {
      const char* g = (const char*)(Bt + (size_t)(bn + smr) * GSB + k0) + sco;
      char* l = (char*)&sB[buf][smr * LS] + sco;
      async_cp16(g, l);
      async_cp16(g + 16, l + 16);
    }
  };

  v8f acc[4][2] = {};

  stage(0, 0);
  wait_async0();
  __syncthreads();

  const int nk = K >> 5;
  for (int ks = 0; ks < nk; ++ks) {
    const int cur = ks & 1;
    if (ks + 1 < nk) stage(1 - cur, (ks + 1) * 32);   // prefetch next slab

    AFragU af[4], bfr[2];
#pragma unroll
    for (int mi = 0; mi < 4; ++mi) {
      const int m = wm + mi * 16 + l16;
#pragma unroll
      for (int e = 0; e < 8; ++e)
        af[mi].u[e] = *(const unsigned*)&sA[cur][m * LS + amapA(e, hi)];
    }
#pragma unroll
    for (int ni = 0; ni < 2; ++ni) {
      const int n = wn + ni * 16 + l16;
#pragma unroll
      for (int e = 0; e < 8; ++e)
        bfr[ni].u[e] = *(const unsigned*)&sB[cur][n * LS + amapB(e, hi)];
    }
#pragma unroll
    for (int mi = 0; mi < 4; ++mi)
#pragma unroll
      for (int ni = 0; ni < 2; ++ni)
        acc[mi][ni] = __builtin_amdgcn_wmma_f32_16x16x32_bf16(
            false, af[mi].v, false, bfr[ni].v, (short)0, acc[mi][ni], false, false);

    wait_async0();       // own prefetches landed
    __syncthreads();     // everyone done reading cur / writing next
  }

  // Epilogue: C/D layout -> lanes 0-15: M=r, N=lane; lanes 16-31: M=8+r.
#pragma unroll
  for (int mi = 0; mi < 4; ++mi)
#pragma unroll
    for (int ni = 0; ni < 2; ++ni) {
      const int nG = bn + wn + ni * 16 + l16;
      const float bv = bias[nG];
#pragma unroll
      for (int r = 0; r < 8; ++r) {
        const int mG = bm + wm + mi * 16 + hi * 8 + r;
        const float val = acc[mi][ni][r] + bv;
        if (C_BF16) ((__bf16*)Craw)[(size_t)mG * N + nG] = (__bf16)val;
        else        ((float*) Craw)[(size_t)mG * N + nG] = val;
      }
    }
}

// ---------------------------------------------------------------------------
// Prep kernels: one-time f32 -> bf16 conversions / transposes.
// ---------------------------------------------------------------------------
__global__ __launch_bounds__(256) void cvt_f32_to_bf16_x4(
    const float* __restrict__ in, __bf16* __restrict__ out, long n4)
{
  const long id = (long)blockIdx.x * 256 + threadIdx.x;
  if (id >= n4) return;
  const float4 f = ((const float4*)in)[id];
  out[id * 4 + 0] = (__bf16)f.x;
  out[id * 4 + 1] = (__bf16)f.y;
  out[id * 4 + 2] = (__bf16)f.z;
  out[id * 4 + 3] = (__bf16)f.w;
}

__global__ __launch_bounds__(256) void transpose_to_bf16(
    const float* __restrict__ w, __bf16* __restrict__ wt, int K, int N, int GS)
{
  const int id = blockIdx.x * 256 + threadIdx.x;   // over K*N
  const int k = id % K;
  const int n = id / K;
  wt[(size_t)n * GS + k] = (__bf16)w[(size_t)k * N + n];
}

__global__ __launch_bounds__(256) void cvt_cb_pad(
    const float* __restrict__ cb, __bf16* __restrict__ cbp)
{
  const int id = blockIdx.x * 256 + threadIdx.x;   // 1024*128
  const int k = id & 127, c = id >> 7;
  cbp[(size_t)c * 136 + k] = (__bf16)cb[id];
}

// Apply BN scale/shift + ReLU elementwise: h_act = relu(h_pre*scl + sft), bf16.
__global__ __launch_bounds__(256) void bn_relu_act(
    const __bf16* __restrict__ hp, const float* __restrict__ scl,
    const float* __restrict__ sft, __bf16* __restrict__ ha, int cmask, long total)
{
  const long id = (long)blockIdx.x * 256 + threadIdx.x;
  if (id >= total) return;
  const int col = (int)(id & cmask);
  float f = (float)hp[id] * scl[col] + sft[col];
  ha[id] = (__bf16)(f > 0.f ? f : 0.f);
}

// ---------------------------------------------------------------------------
// Deterministic column mean/var (training-mode BN): two-level reduction.
// ---------------------------------------------------------------------------
template <typename T>
__global__ __launch_bounds__(256) void col_stats_partial(
    const T* __restrict__ in, float* __restrict__ ps, float* __restrict__ pq,
    int M, int C)
{
  const int col = blockIdx.x * 256 + threadIdx.x;
  const int seg = blockIdx.y;            // 128 segments
  const int rps = M >> 7;
  const T* p = in + (size_t)seg * rps * C + col;
  float s = 0.f, q = 0.f;
  for (int i = 0; i < rps; ++i) {
    const float v = (float)p[(size_t)i * C];
    s += v; q += v * v;
  }
  ps[seg * C + col] = s;
  pq[seg * C + col] = q;
}

__global__ __launch_bounds__(256) void col_stats_final(
    const float* __restrict__ ps, const float* __restrict__ pq,
    const float* __restrict__ gamma, const float* __restrict__ beta,
    float* __restrict__ scl, float* __restrict__ sft, int M, int C)
{
  const int col = blockIdx.x * 256 + threadIdx.x;
  float s = 0.f, q = 0.f;
  for (int i = 0; i < 128; ++i) { s += ps[i * C + col]; q += pq[i * C + col]; }
  const float inv  = 1.f / (float)M;
  const float mean = s * inv;
  const float var  = q * inv - mean * mean;     // biased, matches jnp.var
  const float sc   = gamma[col] * rsqrtf(var + 1e-5f);
  scl[col] = sc;
  sft[col] = beta[col] - mean * sc;
}

__global__ __launch_bounds__(256) void codebook_halfsq(
    const float* __restrict__ cb, float* __restrict__ hsq)
{
  const int c = blockIdx.x * 256 + threadIdx.x;  // 1024 codes
  const float* p = cb + (size_t)c * 128;
  float s = 0.f;
  for (int k = 0; k < 128; ++k) { const float v = p[k]; s += v * v; }
  hsq[c] = 0.5f * s;
}

// ---------------------------------------------------------------------------
// VQ: per 128 reshaped rows, WMMA scores vs all 1024 codes (8 tiles of 128),
// codebook tiles async-copied from padded bf16, running argmax in registers,
// then out = z + (codebook[idx] - z) in f32. Score LDS aliases the codebook
// tile (disjoint in time).
// ---------------------------------------------------------------------------
__global__ __launch_bounds__(256) void vq_argmin_wmma(
    const float* __restrict__ zpre, const float* __restrict__ scl,
    const float* __restrict__ sft, const __bf16* __restrict__ cbp,
    const float* __restrict__ cb, const float* __restrict__ hsq,
    float* __restrict__ out)
{
  constexpr int KS  = 132;                      // z tile K-stride (bf16)
  constexpr int KSB = 136;                      // codebook K-stride (272B rows)
  __shared__ __align__(16) __bf16 sA[128 * KS];         // 33792 B
  __shared__ __align__(16) float  sScores[128 * 128];   // 65536 B, aliased w/ sB
  __bf16* sB = (__bf16*)sScores;                        // 128*136*2 = 34816 B
  __shared__ float bestV[256];
  __shared__ int   bestI[256];

  const int tid  = threadIdx.x;
  const int lane = tid & 31;
  const int wv   = tid >> 5;
  const int l16  = lane & 15;
  const int hi   = lane >> 4;
  const int wm   = (wv >> 2) * 64;
  const int wn   = (wv & 3) * 32;
  const int row0 = blockIdx.x * 128;

  const int rr = tid >> 1;             // staging/scan row or code
  const int kh = (tid & 1) * 64;       // K half

  // Stage BN-applied z tile once (reshaped row m -> zpre[m>>2][(m&3)*128 + k]).
  {
    const int m = row0 + rr;
    const int orow = m >> 2;
    const int c0   = (m & 3) * 128;
    const float* zp = zpre + (size_t)orow * 512 + c0;
#pragma unroll 4
    for (int k = kh; k < kh + 64; ++k) {
      const int col = c0 + k;
      sA[rr * KS + k] = (__bf16)(zp[k] * scl[col] + sft[col]);
    }
  }

  float bv = -3.4e38f;
  int   bi = 0;

  for (int ct = 0; ct < 8; ++ct) {
    __syncthreads();                            // scores/sB reuse fence
    {
      const char* g = (const char*)(cbp + (size_t)(ct * 128 + rr) * KSB + kh);
      char* l = (char*)(sB + rr * KSB + kh);
#pragma unroll
      for (int j = 0; j < 8; ++j) async_cp16(g + j * 16, l + j * 16);
    }
    wait_async0();
    __syncthreads();

    v8f acc[4][2] = {};
#pragma unroll
    for (int ks = 0; ks < 4; ++ks) {
      const int k0 = ks * 32;
      AFragU af[4], bfr[2];
#pragma unroll
      for (int mi = 0; mi < 4; ++mi) {
        const int m = wm + mi * 16 + l16;
#pragma unroll
        for (int e = 0; e < 8; ++e)
          af[mi].u[e] = *(const unsigned*)&sA[m * KS + k0 + amapA(e, hi)];
      }
#pragma unroll
      for (int ni = 0; ni < 2; ++ni) {
        const int n = wn + ni * 16 + l16;
#pragma unroll
        for (int e = 0; e < 8; ++e)
          bfr[ni].u[e] = *(const unsigned*)&sB[n * KSB + k0 + amapB(e, hi)];
      }
#pragma unroll
      for (int mi = 0; mi < 4; ++mi)
#pragma unroll
        for (int ni = 0; ni < 2; ++ni)
          acc[mi][ni] = __builtin_amdgcn_wmma_f32_16x16x32_bf16(
              false, af[mi].v, false, bfr[ni].v, (short)0, acc[mi][ni], false, false);
    }
    __syncthreads();                            // all waves done reading sB
#pragma unroll
    for (int mi = 0; mi < 4; ++mi)
#pragma unroll
      for (int ni = 0; ni < 2; ++ni) {
        const int n = wn + ni * 16 + l16;
#pragma unroll
        for (int r = 0; r < 8; ++r)
          sScores[(wm + mi * 16 + hi * 8 + r) * 128 + n] = acc[mi][ni][r];
      }
    __syncthreads();
    {
      const float* sr = &sScores[rr * 128 + kh];
      const float* hq = &hsq[ct * 128 + kh];
#pragma unroll 4
      for (int j = 0; j < 64; ++j) {
        const float v = sr[j] - hq[j];          // max(z.c - 0.5||c||^2) == argmin d
        if (v > bv) { bv = v; bi = ct * 128 + kh + j; }
      }
    }
  }

  bestV[tid] = bv;
  bestI[tid] = bi;
  __syncthreads();

  {
    const float v0 = bestV[rr * 2], v1 = bestV[rr * 2 + 1];
    const int idx = (v1 > v0) ? bestI[rr * 2 + 1] : bestI[rr * 2];  // ties -> lower idx
    const int m  = row0 + rr;
    const int orow = m >> 2;
    const int c0   = (m & 3) * 128;
    const float* cp = cb + (size_t)idx * 128;
    const float* zp = zpre + (size_t)orow * 512 + c0;
    float* op = out + (size_t)orow * 512 + c0;
#pragma unroll 4
    for (int k = kh; k < kh + 64; ++k) {
      const int col = c0 + k;
      const float z = zp[k] * scl[col] + sft[col];
      op[k] = z + (cp[k] - z);                  // straight-through forward value
    }
  }
}

// ---------------------------------------------------------------------------
extern "C" void kernel_launch(void* const* d_in, const int* in_sizes, int n_in,
                              void* d_out, int out_size, void* d_ws, size_t ws_size,
                              hipStream_t stream)
{
  (void)in_sizes; (void)n_in; (void)out_size; (void)ws_size;

  const float* x   = (const float*)d_in[0];
  const float* w1  = (const float*)d_in[1];
  const float* b1  = (const float*)d_in[2];
  const float* g1  = (const float*)d_in[3];
  const float* be1 = (const float*)d_in[4];
  const float* w2  = (const float*)d_in[5];
  const float* b2  = (const float*)d_in[6];
  const float* g2  = (const float*)d_in[7];
  const float* be2 = (const float*)d_in[8];
  const float* cbk = (const float*)d_in[9];
  // d_in[10..17] (decoder params) are dead in the reference's returned value.

  char* ws = (char*)d_ws;
  size_t off = 0;
  auto alloc = [&](size_t bytes) -> void* {
    void* p = ws + off;
    off += (bytes + 255) & ~(size_t)255;
    return p;
  };
  __bf16* h_pre = (__bf16*)alloc((size_t)32768 * 1024 * 2);  // GEMM1 out (pre-BN)
  float*  z_pre = (float*) alloc((size_t)32768 * 512 * 4);   // GEMM2 out (pre-BN)
  __bf16* h_act = (__bf16*)alloc((size_t)32768 * 1024 * 2);  // relu(bn1(h))
  __bf16* xb    = (__bf16*)alloc((size_t)32768 * 512 * 2);   // x in bf16
  __bf16* w1t   = (__bf16*)alloc((size_t)1024 * 520 * 2);    // w1^T, padded rows
  __bf16* w2t   = (__bf16*)alloc((size_t)512 * 1032 * 2);    // w2^T, padded rows
  __bf16* cbp   = (__bf16*)alloc((size_t)1024 * 136 * 2);    // codebook, padded
  float*  pS    = (float*) alloc((size_t)128 * 1024 * 4);
  float*  pQ    = (float*) alloc((size_t)128 * 1024 * 4);
  float*  scl1  = (float*) alloc(1024 * 4);
  float*  sft1  = (float*) alloc(1024 * 4);
  float*  scl2  = (float*) alloc(512 * 4);
  float*  sft2  = (float*) alloc(512 * 4);
  float*  hsq   = (float*) alloc(1024 * 4);

  // ---- one-time operand prep (bf16 conversions / transposes) ----
  cvt_f32_to_bf16_x4<<<(32768 * 512 / 4) / 256, 256, 0, stream>>>(x, xb, 32768L * 512 / 4);
  transpose_to_bf16<<<(512 * 1024) / 256, 256, 0, stream>>>(w1, w1t, 512, 1024, 520);
  transpose_to_bf16<<<(1024 * 512) / 256, 256, 0, stream>>>(w2, w2t, 1024, 512, 1032);
  cvt_cb_pad<<<(1024 * 128) / 256, 256, 0, stream>>>(cbk, cbp);
  codebook_halfsq<<<4, 256, 0, stream>>>(cbk, hsq);

  // ---- GEMM1: h_pre(bf16) = x @ w1 + b1 ----
  gemm_wmma_async<true><<<dim3(256, 8), 256, 0, stream>>>(
      xb, w1t, b1, h_pre, 32768, 512, 1024, 520);
  // BN1 batch stats -> scale/shift, then fused BN+ReLU into bf16 activations
  col_stats_partial<__bf16><<<dim3(4, 128), 256, 0, stream>>>(h_pre, pS, pQ, 32768, 1024);
  col_stats_final<<<4, 256, 0, stream>>>(pS, pQ, g1, be1, scl1, sft1, 32768, 1024);
  bn_relu_act<<<(32768L * 1024) / 256, 256, 0, stream>>>(
      h_pre, scl1, sft1, h_act, 1023, 32768L * 1024);

  // ---- GEMM2: z_pre(f32) = h_act @ w2 + b2 ----
  gemm_wmma_async<false><<<dim3(256, 4), 256, 0, stream>>>(
      h_act, w2t, b2, z_pre, 32768, 1024, 512, 1032);
  // BN2 batch stats
  col_stats_partial<float><<<dim3(2, 128), 256, 0, stream>>>(z_pre, pS, pQ, 32768, 512);
  col_stats_final<<<2, 256, 0, stream>>>(pS, pQ, g2, be2, scl2, sft2, 32768, 512);

  // ---- VQ: nearest code + straight-through output ----
  vq_argmin_wmma<<<1024, 256, 0, stream>>>(
      z_pre, scl2, sft2, cbp, cbk, hsq, (float*)d_out);
}
